// MultiLatentAttention_71090298683496
// MI455X (gfx1250) — matrix operand order repaired and checked by reading, not compiled, not run on hardware
//
#include <hip/hip_runtime.h>
#include <stdint.h>

// ---------------- problem constants ----------------
#define BATCH   2
#define S_LEN   2048
#define HID     2048
#define NHH     16
#define DN_     128
#define DR_     64
#define DV_     128
#define RANK_   512
#define DQK_    192          // DN + DR
#define NQ      3072         // NH*DQK
#define NKVA    576          // RANK + DR
#define NKVB    4096         // NH*(DN+DV)
#define ROWS    (BATCH*S_LEN) // 4096
#define SCALE_  0.07216878364870323f  // 192^-0.5
#define EPS_    1e-5f

typedef unsigned short u16;
typedef __attribute__((ext_vector_type(16))) __bf16 v16bf;
typedef __attribute__((ext_vector_type(8)))  float  v8f;

union Frag { v16bf v; uint32_t u[8]; };

__device__ __forceinline__ u16 bf16_of(float f) {
    uint32_t x = __float_as_uint(f);
    uint32_t r = (x + 0x7FFFu + ((x >> 16) & 1u)) >> 16;  // RNE
    return (u16)r;
}

// ---------------- fp32 -> bf16 convert ----------------
__global__ __launch_bounds__(256)
void k_cvt_bf16(const float* __restrict__ src, u16* __restrict__ dst, size_t n) {
    for (size_t i = (size_t)blockIdx.x * blockDim.x + threadIdx.x; i < n;
         i += (size_t)gridDim.x * blockDim.x)
        dst[i] = bf16_of(__builtin_nontemporal_load(src + i));
}

// ---------------- generic bf16 WMMA GEMM: C(f32)[M,N] = A[M,K] * B[K,N] ----------------
// block = 256 threads (8 waves); block tile 128x64; K-step 32; each wave: 2x2 16x16 tiles.
// A tile staged with GLOBAL_LOAD_ASYNC_TO_LDS_B128 (ASYNCcnt path); B tile staged
// through VGPRs because it is transposed on the way into LDS.
__global__ __launch_bounds__(256)
void k_gemm_bf16(const u16* __restrict__ A, const u16* __restrict__ B,
                 float* __restrict__ C, int M, int N, int K) {
    __shared__ __align__(16) u16 ldsA[128][40];   // [m][k], 80B row stride (16B aligned)
    __shared__ __align__(16) u16 ldsB[64][40];    // transposed [n][k]

    const int tid  = threadIdx.x;
    const int wave = tid >> 5, lane = tid & 31;
    const int hi = (lane >> 4) & 1, ln = lane & 15;
    const int waveM = wave >> 1, waveN = wave & 1;
    const size_t bm = (size_t)blockIdx.y * 128, bn = (size_t)blockIdx.x * 64;

    // per-thread staging coordinates (fixed across k-steps)
    const int aRow = tid >> 1, aHalf = (tid & 1) * 16;           // A: 32B chunk
    const int bKr  = tid >> 3, bNb   = (tid & 7) * 8;            // B: 16B chunk
    const uint32_t aLds = (uint32_t)(uintptr_t)&ldsA[aRow][aHalf]; // low 32 bits of a
                                        // generic __shared__ pointer == wave LDS offset

    v8f zero = {};
    v8f acc[2][2];
    acc[0][0] = zero; acc[0][1] = zero; acc[1][0] = zero; acc[1][1] = zero;

    for (int k0 = 0; k0 < K; k0 += 32) {
        {   // stage A tile 128x32 asynchronously: global -> LDS, no VGPR round trip.
            // IOFFSET is added to BOTH the LDS and global address (ISA 08 pseudocode),
            // so one address pair covers both 16B chunks of this thread's 32B slice.
            uint64_t aSrc = (uint64_t)(uintptr_t)(A + (bm + aRow) * (size_t)K + k0 + aHalf);
            asm volatile("global_load_async_to_lds_b128 %0, %1, off"
                         :: "v"(aLds), "v"(aSrc) : "memory");
            asm volatile("global_load_async_to_lds_b128 %0, %1, off offset:16"
                         :: "v"(aLds), "v"(aSrc) : "memory");
        }
        {   // stage B tile 32x64 transposed -> ldsB[n][k]
            uint4 w = *(const uint4*)(B + (size_t)(k0 + bKr) * N + bn + bNb);
            if (k0 + 32 < K)  // hint next B tile toward L2
                __builtin_prefetch(B + (size_t)(k0 + 32 + bKr) * N + bn + bNb, 0, 1);
            const u16* ws = (const u16*)&w;
            #pragma unroll
            for (int x = 0; x < 8; ++x) ldsB[bNb + x][bKr] = ws[x];
        }
        asm volatile("s_wait_asynccnt 0x0" ::: "memory"); // A tile landed in LDS
        __syncthreads();                                   // (compiler waits DScnt for B)

        Frag Af[2], Bf[2];
        #pragma unroll
        for (int mi = 0; mi < 2; ++mi) {           // A 16x32 fragment (pairs layout)
            int row = waveM * 32 + mi * 16 + ln;
            #pragma unroll
            for (int j = 0; j < 8; ++j) {
                int k = 16 * hi + 2 * j;
                Af[mi].u[j] = *(const uint32_t*)&ldsA[row][k];
            }
        }
        #pragma unroll
        for (int ni = 0; ni < 2; ++ni) {           // B 32x16 fragment: K contiguous
            int col = waveN * 32 + ni * 16 + ln;
            const uint32_t* p = (const uint32_t*)&ldsB[col][16 * hi];
            #pragma unroll
            for (int j = 0; j < 8; ++j) Bf[ni].u[j] = p[j];
        }
        #pragma unroll
        for (int mi = 0; mi < 2; ++mi)
            #pragma unroll
            for (int ni = 0; ni < 2; ++ni)
                acc[mi][ni] = __builtin_amdgcn_wmma_f32_16x16x32_bf16(
                    false, Af[mi].v, false, Bf[ni].v, (short)0, acc[mi][ni], false, false);
        __syncthreads();
    }

    #pragma unroll
    for (int mi = 0; mi < 2; ++mi)
        #pragma unroll
        for (int ni = 0; ni < 2; ++ni)
            #pragma unroll
            for (int j = 0; j < 8; ++j) {
                size_t row = bm + waveM * 32 + mi * 16 + j + 8 * hi;
                size_t col = bn + waveN * 32 + ni * 16 + ln;
                // write-once stream; keep it out of the near caches
                __builtin_nontemporal_store(acc[mi][ni][j], C + row * N + col);
            }
}

// ---------------- q: copy nope + RoPE on rope part, -> bf16 [row][h][192] ----------------
__global__ __launch_bounds__(256)
void k_make_q(const float* __restrict__ qf, const float* __restrict__ freqs,
              u16* __restrict__ qb) {
    size_t n = (size_t)ROWS * NHH * 96;
    for (size_t i = (size_t)blockIdx.x * blockDim.x + threadIdx.x; i < n;
         i += (size_t)gridDim.x * blockDim.x) {
        size_t rh = i / 96; int p = (int)(i % 96);
        size_t base = rh * DQK_;
        if (p < 64) {
            qb[base + 2 * p]     = bf16_of(qf[base + 2 * p]);
            qb[base + 2 * p + 1] = bf16_of(qf[base + 2 * p + 1]);
        } else {
            int pr = p - 64;
            int t  = (int)((rh >> 4) % S_LEN);
            float th = freqs[(size_t)t * (DR_ / 2) + pr];
            float c = __cosf(th), sn = __sinf(th);
            float x1 = qf[base + DN_ + 2 * pr], x2 = qf[base + DN_ + 2 * pr + 1];
            qb[base + DN_ + 2 * pr]     = bf16_of(x1 * c - x2 * sn);
            qb[base + DN_ + 2 * pr + 1] = bf16_of(x1 * sn + x2 * c);
        }
    }
}

// ---------------- kv: LayerNorm(512) -> bf16, RoPE(k_rope 64) -> bf16 ----------------
__global__ __launch_bounds__(256)
void k_make_kv(const float* __restrict__ kv, const float* __restrict__ gamma,
               const float* __restrict__ beta, const float* __restrict__ freqs,
               u16* __restrict__ kvc, u16* __restrict__ krope) {
    __shared__ float red0[256], red1[256];
    const int row = blockIdx.x, tid = threadIdx.x;
    const int t = row % S_LEN;
    const float* x = kv + (size_t)row * NKVA;
    float a = x[tid], b = x[256 + tid];
    red0[tid] = a + b; red1[tid] = a * a + b * b;
    __syncthreads();
    for (int off = 128; off; off >>= 1) {
        if (tid < off) { red0[tid] += red0[tid + off]; red1[tid] += red1[tid + off]; }
        __syncthreads();
    }
    float mu  = red0[0] * (1.0f / RANK_);
    float var = red1[0] * (1.0f / RANK_) - mu * mu;
    float inv = rsqrtf(var + EPS_);
    kvc[(size_t)row * RANK_ + tid]       = bf16_of((a - mu) * inv * gamma[tid] + beta[tid]);
    kvc[(size_t)row * RANK_ + 256 + tid] = bf16_of((b - mu) * inv * gamma[256 + tid] + beta[256 + tid]);
    if (tid < 32) {
        float th = freqs[(size_t)t * 32 + tid];
        float c = __cosf(th), sn = __sinf(th);
        float x1 = x[RANK_ + 2 * tid], x2 = x[RANK_ + 2 * tid + 1];
        krope[(size_t)row * DR_ + 2 * tid]     = bf16_of(x1 * c - x2 * sn);
        krope[(size_t)row * DR_ + 2 * tid + 1] = bf16_of(x1 * sn + x2 * c);
    }
}

// ---------------- assemble K = [k_nope | k_rope broadcast], V from kvb ----------------
__global__ __launch_bounds__(256)
void k_build_k(const float* __restrict__ kvb, const u16* __restrict__ krope,
               u16* __restrict__ Kb) {
    size_t n = (size_t)ROWS * NHH * DQK_;
    for (size_t i = (size_t)blockIdx.x * blockDim.x + threadIdx.x; i < n;
         i += (size_t)gridDim.x * blockDim.x) {
        size_t rh = i / DQK_; int d = (int)(i % DQK_);
        size_t row = rh >> 4; int h = (int)(rh & 15);
        Kb[i] = (d < DN_) ? bf16_of(kvb[row * NKVB + h * (DN_ + DV_) + d])
                          : krope[row * DR_ + (d - DN_)];
    }
}
__global__ __launch_bounds__(256)
void k_build_v(const float* __restrict__ kvb, u16* __restrict__ Vb) {
    size_t n = (size_t)ROWS * NHH * DV_;
    for (size_t i = (size_t)blockIdx.x * blockDim.x + threadIdx.x; i < n;
         i += (size_t)gridDim.x * blockDim.x) {
        size_t rh = i / DV_; int d = (int)(i % DV_);
        size_t row = rh >> 4; int h = (int)(rh & 15);
        Vb[i] = bf16_of(kvb[row * NKVB + h * (DN_ + DV_) + DN_ + d]);
    }
}

// ---------------- flash attention: grid (S/64, NH, B), 4 waves; wave = 16 t-rows ----------------
__global__ __launch_bounds__(128)
void k_attn(const u16* __restrict__ Q, const u16* __restrict__ Kb,
            const u16* __restrict__ Vb, u16* __restrict__ O) {
    __shared__ __align__(16) u16 ldsVT[128][34];     // V tile transposed [d][s]
    __shared__ __align__(16) u16 ldsP[4][16][32];    // per-wave P tile [t][s]

    const int tid  = threadIdx.x;
    const int wave = tid >> 5, lane = tid & 31;
    const int hi = (lane >> 4) & 1, ln = lane & 15;
    const int h = blockIdx.y, b = blockIdx.z;
    const int tWave = blockIdx.x * 64 + wave * 16;

    // preload Q A-fragments for d = 0..191 (6 fragments), row M = ln
    const u16* qRow = Q + (((size_t)b * S_LEN + tWave + ln) * NHH + h) * DQK_;
    Frag Aq[6];
    #pragma unroll
    for (int f = 0; f < 6; ++f)
        #pragma unroll
        for (int j = 0; j < 8; ++j) {
            int d = 32 * f + 16 * hi + 2 * j;
            Aq[f].u[j] = *(const uint32_t*)(qRow + d);
        }

    v8f zero = {};
    v8f oacc[8];
    float m[8], l[8];
    #pragma unroll
    for (int j = 0; j < 8; ++j) { m[j] = -1e30f; l[j] = 0.f; }
    #pragma unroll
    for (int dt = 0; dt < 8; ++dt) oacc[dt] = zero;

    for (int s0 = 0; s0 < S_LEN; s0 += 32) {
        {   // stage V tile 32x128 transposed into LDS
            int sr = tid >> 2, db = (tid & 3) * 32;
            const u16* vRow = Vb + (((size_t)b * S_LEN + s0 + sr) * NHH + h) * DV_ + db;
            #pragma unroll
            for (int x = 0; x < 32; ++x) ldsVT[db + x][sr] = vRow[x];
        }
        __syncthreads();

        // scores: two 16x16 tiles along s (nt), chained over d
        v8f sc[2];
        #pragma unroll
        for (int nt = 0; nt < 2; ++nt) {
            v8f acc = zero;
            const u16* kRow = Kb + (((size_t)b * S_LEN + s0 + nt * 16 + ln) * NHH + h) * DQK_;
            #pragma unroll
            for (int f = 0; f < 6; ++f) {
                Frag Bk;
                const uint32_t* p = (const uint32_t*)(kRow + 32 * f + 16 * hi);
                #pragma unroll
                for (int j = 0; j < 8; ++j) Bk.u[j] = p[j];
                acc = __builtin_amdgcn_wmma_f32_16x16x32_bf16(
                    false, Aq[f].v, false, Bk.v, (short)0, acc, false, false);
            }
            sc[nt] = acc;
        }

        // online softmax per row (rows j / j+8 live in 16-lane halves)
        float alpha[8];
        #pragma unroll
        for (int j = 0; j < 8; ++j) {
            float a0 = sc[0][j] * SCALE_, a1 = sc[1][j] * SCALE_;
            float mx = fmaxf(a0, a1);
            #pragma unroll
            for (int off = 1; off < 16; off <<= 1) mx = fmaxf(mx, __shfl_xor(mx, off, 32));
            float mn = fmaxf(m[j], mx);
            float al = __expf(m[j] - mn);
            float p0 = __expf(a0 - mn), p1 = __expf(a1 - mn);
            sc[0][j] = p0; sc[1][j] = p1;
            float rs = p0 + p1;
            #pragma unroll
            for (int off = 1; off < 16; off <<= 1) rs += __shfl_xor(rs, off, 32);
            l[j] = l[j] * al + rs;
            m[j] = mn; alpha[j] = al;
        }
        #pragma unroll
        for (int dt = 0; dt < 8; ++dt)
            #pragma unroll
            for (int j = 0; j < 8; ++j) oacc[dt][j] = oacc[dt][j] * alpha[j];

        // C-layout -> A-layout re-tiling of P via per-wave LDS (DS ops are in-order per wave)
        #pragma unroll
        for (int j = 0; j < 8; ++j) {
            int row = j + 8 * hi;
            ldsP[wave][row][ln]      = bf16_of(sc[0][j]);
            ldsP[wave][row][16 + ln] = bf16_of(sc[1][j]);
        }
        asm volatile("" ::: "memory");   // keep compiler from reordering LDS RAW

        Frag Pa;
        #pragma unroll
        for (int j = 0; j < 8; ++j) {
            int k = 16 * hi + 2 * j;
            Pa.u[j] = *(const uint32_t*)&ldsP[wave][ln][k];
        }
        #pragma unroll
        for (int dt = 0; dt < 8; ++dt) {
            Frag Vf;
            const uint32_t* p = (const uint32_t*)&ldsVT[dt * 16 + ln][16 * hi];
            #pragma unroll
            for (int j = 0; j < 8; ++j) Vf.u[j] = p[j];
            oacc[dt] = __builtin_amdgcn_wmma_f32_16x16x32_bf16(
                false, Pa.v, false, Vf.v, (short)0, oacc[dt], false, false);
        }
        __syncthreads();
    }

    // epilogue: normalize by l, write bf16 attn [row][h*128+d]
    #pragma unroll
    for (int dt = 0; dt < 8; ++dt)
        #pragma unroll
        for (int j = 0; j < 8; ++j) {
            int t = tWave + j + 8 * hi;
            int d = dt * 16 + ln;
            O[((size_t)b * S_LEN + t) * (NHH * DV_) + h * DV_ + d] = bf16_of(oacc[dt][j] / l[j]);
        }
}

// ---------------- host-side orchestration ----------------
extern "C" void kernel_launch(void* const* d_in, const int* in_sizes, int n_in,
                              void* d_out, int out_size, void* d_ws, size_t ws_size,
                              hipStream_t stream) {
    (void)in_sizes; (void)n_in; (void)out_size; (void)ws_size;
    const float* hid   = (const float*)d_in[0];
    const float* freqs = (const float*)d_in[1];
    const float* Wq    = (const float*)d_in[2];
    const float* Wkva  = (const float*)d_in[3];
    const float* gamma = (const float*)d_in[4];
    const float* beta  = (const float*)d_in[5];
    const float* Wkvb  = (const float*)d_in[6];
    const float* Wout  = (const float*)d_in[7];
    float* out = (float*)d_out;

    char* w = (char*)d_ws;
    auto alloc = [&](size_t bytes) -> char* {
        char* p = w; w += (bytes + 255) & ~(size_t)255; return p;
    };
    u16*   hidB   = (u16*)  alloc((size_t)ROWS * HID * 2);
    u16*   WqB    = (u16*)  alloc((size_t)HID * NQ * 2);
    u16*   WkvaB  = (u16*)  alloc((size_t)HID * NKVA * 2);
    u16*   WkvbB  = (u16*)  alloc((size_t)RANK_ * NKVB * 2);
    u16*   WoutB  = (u16*)  alloc((size_t)(NHH * DV_) * HID * 2);
    // big region shared by q_f32 (dead after k_make_q) and kvb_f32 (written later)
    char*  bigRgn =         alloc((size_t)ROWS * NKVB * 4);   // 67 MB >= q_f32 (50 MB)
    float* qF     = (float*)bigRgn;
    float* kvbF   = (float*)bigRgn;
    float* kvF    = (float*) alloc((size_t)ROWS * NKVA * 4);
    u16*   qB     = (u16*)   alloc((size_t)ROWS * NHH * DQK_ * 2);
    u16*   kvcB   = (u16*)   alloc((size_t)ROWS * RANK_ * 2);
    u16*   kropeB = (u16*)   alloc((size_t)ROWS * DR_ * 2);
    u16*   KB     = (u16*)   alloc((size_t)ROWS * NHH * DQK_ * 2);
    u16*   VB     = (u16*)   alloc((size_t)ROWS * NHH * DV_ * 2);
    u16*   attnB  = hidB;    // hidB dead after the first two GEMMs; reuse (same size)

    // 1) fp32 -> bf16 conversions
    k_cvt_bf16<<<4096, 256, 0, stream>>>(hid,  hidB,  (size_t)ROWS * HID);
    k_cvt_bf16<<<4096, 256, 0, stream>>>(Wq,   WqB,   (size_t)HID * NQ);
    k_cvt_bf16<<<2048, 256, 0, stream>>>(Wkva, WkvaB, (size_t)HID * NKVA);
    k_cvt_bf16<<<2048, 256, 0, stream>>>(Wkvb, WkvbB, (size_t)RANK_ * NKVB);
    k_cvt_bf16<<<4096, 256, 0, stream>>>(Wout, WoutB, (size_t)(NHH * DV_) * HID);

    // 2) q = X @ Wq  (4096 x 2048 x 3072)
    k_gemm_bf16<<<dim3(NQ / 64, ROWS / 128), 256, 0, stream>>>(hidB, WqB, qF, ROWS, NQ, HID);
    // 3) kv = X @ Wkv_a  (4096 x 2048 x 576)
    k_gemm_bf16<<<dim3(NKVA / 64, ROWS / 128), 256, 0, stream>>>(hidB, WkvaB, kvF, ROWS, NKVA, HID);

    // 4) q RoPE -> bf16 ; kv LayerNorm + k_rope RoPE -> bf16
    k_make_q<<<8192, 256, 0, stream>>>(qF, freqs, qB);
    k_make_kv<<<ROWS, 256, 0, stream>>>(kvF, gamma, beta, freqs, kvcB, kropeB);

    // 5) kvb = kv_c @ Wkv_b  (4096 x 512 x 4096)  -- overwrites qF region (now dead)
    k_gemm_bf16<<<dim3(NKVB / 64, ROWS / 128), 256, 0, stream>>>(kvcB, WkvbB, kvbF, ROWS, NKVB, RANK_);

    // 6) assemble K (nope | rope broadcast) and V
    k_build_k<<<8192, 256, 0, stream>>>(kvbF, kropeB, KB);
    k_build_v<<<8192, 256, 0, stream>>>(kvbF, VB);

    // 7) flash attention -> attnB (bf16, [row][NH*DV]) -- reuses hidB region (now dead)
    k_attn<<<dim3(S_LEN / 64, NHH, BATCH), 128, 0, stream>>>(qB, KB, VB, attnB);

    // 8) out = attn @ Wout  (4096 x 2048 x 2048), fp32 straight to d_out
    k_gemm_bf16<<<dim3(HID / 64, ROWS / 128), 256, 0, stream>>>(attnB, WoutB, out, ROWS, HID, NHH * DV_);
}
